// RNNMemoryCell_77189152244387
// MI455X (gfx1250) — matrix operand order repaired and checked by reading, not compiled, not run on hardware
//
#include <hip/hip_runtime.h>
#include <hip/hip_bf16.h>

#define BATCH 2
#define SEQ   512
#define DIM   512
#define HEADS 8
#define DHEAD 64
#define KSZ   4
#define DI    (HEADS*DHEAD)   // 512
#define BT    (BATCH*SEQ)     // 1024
#define NBH   (BATCH*HEADS)   // 16

typedef __attribute__((ext_vector_type(16))) __bf16 v16bf;
typedef __attribute__((ext_vector_type(8)))  __bf16 v8bf;
typedef __attribute__((ext_vector_type(8)))  float  v8f;

typedef __attribute__((address_space(1))) int gint_t;   // global
typedef __attribute__((address_space(3))) int lint_t;   // LDS

// ---------------------------------------------------------------------------
// CDNA5 async global->LDS copy (ASYNCcnt-tracked) + drain.
// ---------------------------------------------------------------------------
__device__ __forceinline__ void async_copy_f32(const float* g, float* l)
{
#if __has_builtin(__builtin_amdgcn_global_load_async_to_lds_b32)
    __builtin_amdgcn_global_load_async_to_lds_b32(
        (gint_t*)(uintptr_t)g, (lint_t*)(uintptr_t)l, 0, 0);
#else
    // VGLOBAL async encoding: VDST = LDS byte address (generic low 32 bits),
    // VADDR = 64-bit global address (GV mode).
    asm volatile("global_load_async_to_lds_b32 %0, %1, off"
                 :: "v"((unsigned)(uintptr_t)l), "v"(g) : "memory");
#endif
}

__device__ __forceinline__ void wait_async0()
{
#if __has_builtin(__builtin_amdgcn_s_wait_asynccnt)
    __builtin_amdgcn_s_wait_asynccnt(0);
#else
    asm volatile("s_wait_asynccnt 0x0" ::: "memory");
#endif
}

// ---------------------------------------------------------------------------
// Kernel 1: pre-RMSNorm + the 3 sigmoid gates (lr/decay/mom), one block per
// token row. xn kept in LDS for the 24 gate dot-products.
// ---------------------------------------------------------------------------
__global__ void k_rmsnorm_gates(const float* __restrict__ x,
                                const float* __restrict__ pw,
                                const float* __restrict__ lrw, const float* __restrict__ lrb,
                                const float* __restrict__ dew, const float* __restrict__ deb,
                                const float* __restrict__ mow, const float* __restrict__ mob,
                                float* __restrict__ xn, float* __restrict__ gates)
{
    __shared__ float row[DIM];
    __shared__ float red[8];
    const int bt   = blockIdx.x;          // 0..BT-1
    const int tid  = threadIdx.x;         // 0..255
    const int lane = tid & 31, wave = tid >> 5;

    float ss = 0.f;
    for (int i = tid; i < DIM; i += 256) {
        float v = x[bt * DIM + i];
        row[i] = v;
        ss += v * v;
    }
    for (int o = 16; o > 0; o >>= 1) ss += __shfl_xor(ss, o, 32);
    if (lane == 0) red[wave] = ss;
    __syncthreads();
    float tot = 0.f;
    #pragma unroll
    for (int w = 0; w < 8; w++) tot += red[w];
    const float scale = rsqrtf(tot / (float)DIM + 1e-6f);

    for (int i = tid; i < DIM; i += 256) {
        float v = row[i] * scale * pw[i];
        row[i] = v;
        xn[bt * DIM + i] = v;
    }
    __syncthreads();

    const int b = bt / SEQ, t = bt % SEQ;
    for (int c = wave; c < 24; c += 8) {   // 3 gates x 8 heads
        const int g = c >> 3, h = c & 7;
        const float* gw = (g == 0 ? lrw : g == 1 ? dew : mow) + h * DIM;
        const float* gb = (g == 0 ? lrb : g == 1 ? deb : mob);
        float s = 0.f;
        for (int i = lane; i < DIM; i += 32) s += row[i] * gw[i];
        for (int o = 16; o > 0; o >>= 1) s += __shfl_xor(s, o, 32);
        if (lane == 0) {
            float z = s + gb[h];
            gates[((g * BATCH + b) * HEADS + h) * SEQ + t] = 1.f / (1.f + expf(-z));
        }
    }
}

// ---------------------------------------------------------------------------
// Kernel 2: depthwise conv (K=4, pad 2/2, take first T) -> bf16 GEMM inputs
// ---------------------------------------------------------------------------
__global__ void k_dwconv(const float* __restrict__ xn,
                         const float* __restrict__ wq, const float* __restrict__ wk,
                         const float* __restrict__ wv,
                         __bf16* __restrict__ qi, __bf16* __restrict__ ki,
                         __bf16* __restrict__ vi)
{
    const int bt = blockIdx.x;
    const int b = bt / SEQ, t = bt % SEQ;
    for (int c = threadIdx.x; c < DIM; c += 256) {
        float aq = 0.f, ak = 0.f, av = 0.f;
        #pragma unroll
        for (int j = 0; j < KSZ; j++) {
            int tt = t + j - KSZ / 2;
            if (tt >= 0 && tt < SEQ) {
                float v = xn[(b * SEQ + tt) * DIM + c];
                aq += wq[c * KSZ + j] * v;
                ak += wk[c * KSZ + j] * v;
                av += wv[c * KSZ + j] * v;
            }
        }
        qi[bt * DIM + c] = (__bf16)aq;
        ki[bt * DIM + c] = (__bf16)ak;
        vi[bt * DIM + c] = (__bf16)av;
    }
}

// ---------------------------------------------------------------------------
// Kernel 3: f32 -> bf16 weight conversion
// ---------------------------------------------------------------------------
__global__ void k_cvt(const float* __restrict__ src, __bf16* __restrict__ dst, int n)
{
    int i = blockIdx.x * 256 + threadIdx.x;
    if (i < n) dst[i] = (__bf16)src[i];
}

// ---------------------------------------------------------------------------
// Kernel 4: WMMA bf16 GEMM  C[1024x512] = A[1024x512] * W[512x512]^T
// One 16x16 output tile per wave, K-loop of 16 x v_wmma_f32_16x16x32_bf16.
// Fragments use contiguous k = 16*half + s in both A and B (wave32).
// ---------------------------------------------------------------------------
__global__ void k_gemm_bf16(const __bf16* __restrict__ A,
                            const __bf16* __restrict__ W,
                            float* __restrict__ C)
{
    const int wave = threadIdx.x >> 5;
    const int lane = threadIdx.x & 31;
    const int half = lane >> 4;
    const int l16  = lane & 15;
    const int tile = blockIdx.x * 8 + wave;   // 2048 tiles total
    const int tM   = tile >> 5;               // 64 M-tiles
    const int tN   = tile & 31;               // 32 N-tiles

    const int mrow = tM * 16 + l16;
    const int ncol = tN * 16 + l16;

    v8f acc = {};
    #pragma unroll 4
    for (int k0 = 0; k0 < DIM; k0 += 32) {
        v8bf alo = *(const v8bf*)(A + mrow * DIM + k0 + half * 16);
        v8bf ahi = *(const v8bf*)(A + mrow * DIM + k0 + half * 16 + 8);
        v16bf a;
        #pragma unroll
        for (int i = 0; i < 8; i++) { a[i] = alo[i]; a[8 + i] = ahi[i]; }
        v16bf bf = *(const v16bf*)(W + ncol * DIM + k0 + half * 16);
        acc = __builtin_amdgcn_wmma_f32_16x16x32_bf16(
            false, a, false, bf, (short)0, acc, false, false);
    }
    #pragma unroll
    for (int r = 0; r < 8; r++)
        C[(tM * 16 + half * 8 + r) * DIM + tN * 16 + l16] = acc[r];
}

// ---------------------------------------------------------------------------
// Kernel 5: per-head RMSNorm of q and k (gamma-scaled), in place.
// ---------------------------------------------------------------------------
__global__ void k_mhnorm(float* __restrict__ q, float* __restrict__ k,
                         const float* __restrict__ qg, const float* __restrict__ kg)
{
    const int bt   = blockIdx.x;
    const int h    = threadIdx.x >> 5;
    const int lane = threadIdx.x & 31;
    const int base = bt * DI + h * DHEAD;

    float q0 = q[base + lane * 2], q1 = q[base + lane * 2 + 1];
    float k0 = k[base + lane * 2], k1 = k[base + lane * 2 + 1];
    float sq = q0 * q0 + q1 * q1;
    float sk = k0 * k0 + k1 * k1;
    for (int o = 16; o > 0; o >>= 1) {
        sq += __shfl_xor(sq, o, 32);
        sk += __shfl_xor(sk, o, 32);
    }
    float rq = fmaxf(sqrtf(sq) * 0.125f, 1e-8f);   // ||x|| / sqrt(64)
    float rk = fmaxf(sqrtf(sk) * 0.125f, 1e-8f);
    q[base + lane * 2]     = q0 / rq * qg[h * DHEAD + lane * 2];
    q[base + lane * 2 + 1] = q1 / rq * qg[h * DHEAD + lane * 2 + 1];
    k[base + lane * 2]     = k0 / rk * kg[h * DHEAD + lane * 2];
    k[base + lane * 2 + 1] = k1 / rk * kg[h * DHEAD + lane * 2 + 1];
}

// ---------------------------------------------------------------------------
// Kernel 6: momentum-DeltaNet recurrence, register-resident state.
//   u = S k ; y = S q ; M = mom*M + (u-v)k^T ; S = dec*S - lr*M
// Row-parallel: thread (r,p) owns 16 columns of row r of S and M in VGPRs.
// Row dot-products reduce over the 4 owning lanes via two shfl_xor. Shared
// per-step inputs (k,v,q,gates = 195 floats) are double-buffered in LDS and
// prefetched one step ahead with async global->LDS copies (ASYNCcnt), so the
// only per-step sync is one s_wait_asynccnt + one barrier.
// ---------------------------------------------------------------------------
__global__ void k_recurrence(const float* __restrict__ q,
                             const float* __restrict__ k,
                             const float* __restrict__ v,
                             const float* __restrict__ gates,
                             __bf16* __restrict__ y)
{
    __shared__ float buf[2][3 * DHEAD + 4];   // [k:0..63 | v:64..127 | q:128..191 | lr,dec,mom]

    const int bh = blockIdx.x;
    const int b = bh >> 3, h = bh & 7;
    const int tid = threadIdx.x;
    const int r  = tid >> 2;                  // row 0..63
    const int pp = tid & 3;                   // quarter-row 0..3
    const int c0 = pp * 16;

    // Per-thread global prefetch source (element tid of the 195-float step blob)
    const float* gsrc = nullptr;
    int gstride = 0;
    const int tokbase = (b * SEQ) * DI + h * DHEAD;
    if (tid < 64)       { gsrc = k + tokbase + tid;         gstride = DI; }
    else if (tid < 128) { gsrc = v + tokbase + (tid - 64);  gstride = DI; }
    else if (tid < 192) { gsrc = q + tokbase + (tid - 128); gstride = DI; }
    else if (tid < 195) {
        const int g = tid - 192;
        gsrc = gates + ((g * BATCH + b) * HEADS + h) * SEQ;
        gstride = 1;
    }

    float Sr[16], Mr[16];
    #pragma unroll
    for (int i = 0; i < 16; i++) { Sr[i] = 0.f; Mr[i] = 0.f; }

    // Prologue: async-prefetch step 0 into buffer 0.
    if (gsrc) async_copy_f32(gsrc, &buf[0][tid]);

    for (int t = 0; t < SEQ; t++) {
        const int cur = t & 1;
        wait_async0();          // this wave's prefetch landed in LDS
        __syncthreads();        // every wave's prefetch visible

        if ((t + 1) < SEQ && gsrc)
            async_copy_f32(gsrc + (size_t)(t + 1) * gstride, &buf[1 - cur][tid]);

        const float* kb = &buf[cur][0];
        const float* vb = &buf[cur][DHEAD];
        const float* qb = &buf[cur][2 * DHEAD];
        const float lr  = buf[cur][3 * DHEAD + 0];
        const float dec = buf[cur][3 * DHEAD + 1];
        const float mom = buf[cur][3 * DHEAD + 2];

        float u = 0.f, yy = 0.f;
        #pragma unroll
        for (int i = 0; i < 16; i++) {
            float kk = kb[c0 + i], qq = qb[c0 + i];
            u  += Sr[i] * kk;
            yy += Sr[i] * qq;
        }
        u  += __shfl_xor(u, 1, 32);  u  += __shfl_xor(u, 2, 32);
        yy += __shfl_xor(yy, 1, 32); yy += __shfl_xor(yy, 2, 32);

        if (pp == 0) y[tokbase + t * DI + r] = (__bf16)yy;   // y_t = S_{t-1} q_t

        const float a = u - vb[r];
        #pragma unroll
        for (int i = 0; i < 16; i++) {
            float m = mom * Mr[i] + a * kb[c0 + i];
            Mr[i] = m;
            Sr[i] = dec * Sr[i] - lr * m;
        }
        // No trailing barrier: next iteration's wait+barrier orders the next
        // prefetch (which overwrites buf[cur]) after all reads above.
    }
}

// ---------------------------------------------------------------------------
extern "C" void kernel_launch(void* const* d_in, const int* in_sizes, int n_in,
                              void* d_out, int out_size, void* d_ws, size_t ws_size,
                              hipStream_t stream)
{
    const float* x   = (const float*)d_in[0];
    const float* pw  = (const float*)d_in[1];
    const float* cqw = (const float*)d_in[2];
    const float* ckw = (const float*)d_in[3];
    const float* cvw = (const float*)d_in[4];
    const float* w_q = (const float*)d_in[5];
    const float* w_k = (const float*)d_in[6];
    const float* w_v = (const float*)d_in[7];
    const float* w_o = (const float*)d_in[8];
    const float* lrw = (const float*)d_in[9];
    const float* lrb = (const float*)d_in[10];
    const float* dew = (const float*)d_in[11];
    const float* deb = (const float*)d_in[12];
    const float* mow = (const float*)d_in[13];
    const float* mob = (const float*)d_in[14];
    const float* qg  = (const float*)d_in[15];
    const float* kg  = (const float*)d_in[16];

    char* ws = (char*)d_ws;
    size_t off = 0;
    auto alloc = [&](size_t bytes) -> void* {
        void* p = ws + off;
        off += (bytes + 255) & ~(size_t)255;
        return p;
    };
    float*  xn    = (float*) alloc((size_t)BT * DIM * 4);
    float*  gates = (float*) alloc((size_t)3 * BATCH * HEADS * SEQ * 4);
    __bf16* qi    = (__bf16*)alloc((size_t)BT * DIM * 2);
    __bf16* ki    = (__bf16*)alloc((size_t)BT * DIM * 2);
    __bf16* vi    = (__bf16*)alloc((size_t)BT * DIM * 2);
    __bf16* wqb   = (__bf16*)alloc((size_t)DI * DIM * 2);
    __bf16* wkb   = (__bf16*)alloc((size_t)DI * DIM * 2);
    __bf16* wvb   = (__bf16*)alloc((size_t)DI * DIM * 2);
    __bf16* wob   = (__bf16*)alloc((size_t)DIM * DI * 2);
    float*  qf    = (float*) alloc((size_t)BT * DI * 4);
    float*  kf    = (float*) alloc((size_t)BT * DI * 4);
    float*  vf    = (float*) alloc((size_t)BT * DI * 4);
    __bf16* ybf   = (__bf16*)alloc((size_t)BT * DI * 2);
    (void)ws_size; (void)in_sizes; (void)n_in; (void)out_size;

    k_rmsnorm_gates<<<BT, 256, 0, stream>>>(x, pw, lrw, lrb, dew, deb, mow, mob,
                                            xn, gates);
    k_dwconv<<<BT, 256, 0, stream>>>(xn, cqw, ckw, cvw, qi, ki, vi);

    const int nw = DI * DIM;
    k_cvt<<<(nw + 255) / 256, 256, 0, stream>>>(w_q, wqb, nw);
    k_cvt<<<(nw + 255) / 256, 256, 0, stream>>>(w_k, wkb, nw);
    k_cvt<<<(nw + 255) / 256, 256, 0, stream>>>(w_v, wvb, nw);
    k_cvt<<<(nw + 255) / 256, 256, 0, stream>>>(w_o, wob, nw);

    k_gemm_bf16<<<256, 256, 0, stream>>>(qi, wqb, qf);
    k_gemm_bf16<<<256, 256, 0, stream>>>(ki, wkb, kf);
    k_gemm_bf16<<<256, 256, 0, stream>>>(vi, wvb, vf);

    k_mhnorm<<<BT, 256, 0, stream>>>(qf, kf, qg, kg);

    k_recurrence<<<NBH, 256, 0, stream>>>(qf, kf, vf, gates, ybf);

    k_gemm_bf16<<<256, 256, 0, stream>>>(ybf, wob, (float*)d_out);
}
